// PerformerBlock_32066225832681
// MI455X (gfx1250) — compile-verified, matrix operand order checked by reading, not run on hardware
//
#include <hip/hip_runtime.h>
#include <hip/hip_bf16.h>

// ---------------------------------------------------------------------------
// Types for CDNA5 WMMA (wave32): bf16 A/B fragments, f32 accumulator.
// ---------------------------------------------------------------------------
typedef __attribute__((ext_vector_type(16))) __bf16 v16bf;
typedef __attribute__((ext_vector_type(8)))  __bf16 v8bf;
typedef __attribute__((ext_vector_type(2)))  __bf16 v2bf;
typedef __attribute__((ext_vector_type(8)))  float  v8f;

#define DIM      768
#define HEADS    12
#define HEAD_DIM 64
#define MFEAT    384
#define HIDDEN   3072
#define SEQ      1024
#define BATCH    32
#define SCALE_F  0.35355339059327373f      /* 64^-0.25 */
#define INV_SQRT_M 0.05103103630798288f    /* 1/sqrt(384) */

#define BM 128
#define BN 64
#define BK 32
#define LDS_STRIDE 40   /* 40*2B = 80B row stride, multiple of 16B for b128 ds loads */

__device__ __forceinline__ float gelu_tanh(float x) {
    float c = x + 0.044715f * x * x * x;
    return 0.5f * x * (1.0f + tanhf(0.7978845608028654f * c));
}

// ---------------------------------------------------------------------------
// LayerNorm over 768 columns; one block (256 threads) per row.
// ---------------------------------------------------------------------------
__global__ __launch_bounds__(256)
void ln_kernel(const float* __restrict__ x, const float* __restrict__ g,
               const float* __restrict__ b, float* __restrict__ y) {
    __shared__ float sbuf[256];
    const int row = blockIdx.x;
    const int t = threadIdx.x;
    const float* xr = x + (long)row * DIM;
    float v[3];
    float s = 0.f;
#pragma unroll
    for (int i = 0; i < 3; ++i) { v[i] = xr[t + i * 256]; s += v[i]; }
    sbuf[t] = s; __syncthreads();
    for (int o = 128; o > 0; o >>= 1) { if (t < o) sbuf[t] += sbuf[t + o]; __syncthreads(); }
    const float mu = sbuf[0] * (1.0f / DIM);
    __syncthreads();
    float s2 = 0.f;
#pragma unroll
    for (int i = 0; i < 3; ++i) { float d = v[i] - mu; s2 += d * d; }
    sbuf[t] = s2; __syncthreads();
    for (int o = 128; o > 0; o >>= 1) { if (t < o) sbuf[t] += sbuf[t + o]; __syncthreads(); }
    const float rs = rsqrtf(sbuf[0] * (1.0f / DIM) + 1e-5f);
#pragma unroll
    for (int i = 0; i < 3; ++i) {
        int c = t + i * 256;
        y[(long)row * DIM + c] = (v[i] - mu) * rs * g[c] + b[c];
    }
}

// ---------------------------------------------------------------------------
// Per (head, token): 0.5*||x*SCALE||^2 for q and k rows of the qkv buffer.
// 0.5*SCALE^2 = 0.0625.
// ---------------------------------------------------------------------------
__global__ __launch_bounds__(256)
void rowsq_kernel(const float* __restrict__ qkv, float* __restrict__ rq,
                  float* __restrict__ rk) {
    int idx = blockIdx.x * blockDim.x + threadIdx.x;   // < HEADS*SEQ
    int h = idx / SEQ, n = idx % SEQ;
    const float* q = qkv + (long)n * (3 * DIM) + h * HEAD_DIM;
    const float* k = q + DIM;
    float sq = 0.f, sk = 0.f;
#pragma unroll 8
    for (int c = 0; c < HEAD_DIM; ++c) { sq += q[c] * q[c]; sk += k[c] * k[c]; }
    rq[h * SEQ + n] = 0.0625f * sq;
    rk[h * SEQ + n] = 0.0625f * sk;
}

// ---------------------------------------------------------------------------
// k_sum[h][m] = sum_n kf[h][n][m]
// ---------------------------------------------------------------------------
__global__ __launch_bounds__(256)
void ksum_kernel(const float* __restrict__ kf, float* __restrict__ ksum) {
    int idx = blockIdx.x * blockDim.x + threadIdx.x;   // < HEADS*MFEAT
    int h = idx / MFEAT, m = idx % MFEAT;
    const float* base = kf + (long)h * SEQ * MFEAT + m;
    float s = 0.f;
#pragma unroll 8
    for (int n = 0; n < SEQ; ++n) s += base[(long)n * MFEAT];
    ksum[idx] = s;
}

// ---------------------------------------------------------------------------
// dinv[h][n] = 1 / (dot(qf[h][n][:], ksum[h][:]) + eps)
// ---------------------------------------------------------------------------
__global__ __launch_bounds__(256)
void dinv_kernel(const float* __restrict__ qf, const float* __restrict__ ksum,
                 float* __restrict__ dinv) {
    int idx = blockIdx.x * blockDim.x + threadIdx.x;   // < HEADS*SEQ
    int h = idx / SEQ, n = idx % SEQ;
    const float* qr = qf + ((long)h * SEQ + n) * MFEAT;
    const float* ks = ksum + h * MFEAT;
    float d = 0.f;
#pragma unroll 8
    for (int m = 0; m < MFEAT; ++m) d += qr[m] * ks[m];
    dinv[idx] = 1.0f / (d + 1e-8f);
}

// ---------------------------------------------------------------------------
// Generic bf16-WMMA GEMM, 128x64 tile, BK=32, 128 threads (4 waves, wave32).
// Each wave owns a 32x64 strip: 2 A-frags + 4 B-frags -> 8 WMMAs per K-step
// (1.5 ds_load_b128 per WMMA). f32 operands are converted to bf16 in pairs
// (packed v_cvt_pk_bf16_f32 + ds_store_b32) while staging into LDS.
// C[z] = A[z] (rows x K, strides aRow/aCol) @ B[z] (K x N, row stride bRow).
// Epilogues: 0 plain(+bias), 1 exp feature map, 2 scale-by-dinv,
//            3 bias+gelu, 4 bias+residual.
// All dims used here are multiples of the tile sizes -> no bounds checks.
// ---------------------------------------------------------------------------
template <int EPI>
__global__ __launch_bounds__(128)
void wmma_gemm(const float* __restrict__ A, const float* __restrict__ Bmat,
               const float* __restrict__ bias, float* __restrict__ C,
               const float* __restrict__ extra,
               int kdim, int aRow, int aCol, long aZ,
               int bRow, long bZ, int cRow, long cZ,
               long exZ, int exRow, float alpha, float beta) {
    __shared__ __align__(16) __bf16 As[BM][LDS_STRIDE];
    __shared__ __align__(16) __bf16 Bs[BN][LDS_STRIDE];   // stored [n][k]

    const int tid  = threadIdx.x;
    const int lane = tid & 31;
    const int wave = tid >> 5;         // 0..3 : 32-row strip
    const int hi   = (lane >> 4) & 1;  // upper half-wave
    const int l16  = lane & 15;
    const int z = blockIdx.z;

    const float* Ab = A + (long)z * aZ + (long)(blockIdx.y * BM) * aRow;
    const float* Bb = Bmat + (long)z * bZ + blockIdx.x * BN;

    v8f acc[2][4] = {};
    const int nk = kdim / BK;

    for (int kt = 0; kt < nk; ++kt) {
        const int k0 = kt * BK;
        if (kt + 1 < nk) {  // hint next K-tiles into cache (global_prefetch_b8)
            __builtin_prefetch(Ab + (long)tid * aRow + (long)(k0 + BK) * aCol, 0, 1);
            __builtin_prefetch(Bb + (long)(k0 + BK + (tid & 31)) * bRow + (tid >> 5), 0, 1);
        }
        // Stage A tile (BM x BK) as K-pairs: packed f32->bf16 convert, one
        // ds_store_b32 per pair; contiguous b64 global reads when aCol==1.
#pragma unroll
        for (int i = 0; i < 16; ++i) {
            int li = i * 128 + tid;         // pair index, 2048 pairs total
            int r = li >> 4, p = li & 15;   // row 0..127, k-pair 0..15
            const float* ap = Ab + (long)r * aRow + (long)(k0 + 2 * p) * aCol;
            v2bf pr = { (__bf16)ap[0], (__bf16)ap[aCol] };
            *(v2bf*)&As[r][2 * p] = pr;
        }
        // Stage B tile (BK x BN) transposed into Bs[n][k] as K-pairs: two
        // coalesced-b32 global reads (consecutive tids -> consecutive n),
        // one ds_store_b32 per pair.
#pragma unroll
        for (int i = 0; i < 8; ++i) {
            int li = i * 128 + tid;         // pair index, 1024 pairs total
            int n = li & 63, kp = li >> 6;  // n 0..63, k-pair 0..15
            const float* bp = Bb + (long)(k0 + 2 * kp) * bRow + n;
            v2bf pr = { (__bf16)bp[0], (__bf16)bp[bRow] };
            *(v2bf*)&Bs[n][2 * kp] = pr;
        }
        __syncthreads();

        // Load fragments per ISA 16-bit A layout: lanes 0-15 K=0..7,16..23;
        // lanes 16-31 K=8..15,24..31 -> two 16B ds loads each.
        v16bf afr[2], bfr[4];
#pragma unroll
        for (int mf = 0; mf < 2; ++mf) {
            const __bf16* p = &As[wave * 32 + mf * 16 + l16][hi * 8];
            v8bf lo = *(const v8bf*)p;
            v8bf up = *(const v8bf*)(p + 16);
            afr[mf] = __builtin_shufflevector(lo, up, 0,1,2,3,4,5,6,7,8,9,10,11,12,13,14,15);
        }
#pragma unroll
        for (int nf = 0; nf < 4; ++nf) {
            const __bf16* p = &Bs[nf * 16 + l16][hi * 8];
            v8bf lo = *(const v8bf*)p;
            v8bf up = *(const v8bf*)(p + 16);
            bfr[nf] = __builtin_shufflevector(lo, up, 0,1,2,3,4,5,6,7,8,9,10,11,12,13,14,15);
        }
#pragma unroll
        for (int mf = 0; mf < 2; ++mf)
#pragma unroll
            for (int nf = 0; nf < 4; ++nf)
                acc[mf][nf] = __builtin_amdgcn_wmma_f32_16x16x32_bf16(
                    false, afr[mf], false, bfr[nf], (short)0, acc[mf][nf],
                    false, false);
        __syncthreads();
    }

    // Epilogue: C/D VGPR r holds (M = r + 8*hi, N = l16) within each 16x16.
    float* Cb = C + (long)z * cZ;
    const float* exb = extra ? extra + (long)z * exZ : nullptr;
#pragma unroll
    for (int mf = 0; mf < 2; ++mf)
#pragma unroll
        for (int nf = 0; nf < 4; ++nf)
#pragma unroll
            for (int r = 0; r < 8; ++r) {
                int M = wave * 32 + mf * 16 + r + hi * 8;
                int N = nf * 16 + l16;
                int gr = blockIdx.y * BM + M;
                int gc = blockIdx.x * BN + N;
                float v = acc[mf][nf][r];
                if (EPI == 0)      v += bias ? bias[gc] : 0.f;
                else if (EPI == 1) v = __expf(v * alpha - exb[gr]) * beta;
                else if (EPI == 2) v = v * exb[gr];
                else if (EPI == 3) v = gelu_tanh(v + bias[gc]);
                else if (EPI == 4) v = v + bias[gc] + exb[(long)gr * exRow + gc];
                Cb[(long)gr * cRow + gc] = v;
            }
}

// ---------------------------------------------------------------------------
// Host-side orchestration: process one batch index b at a time so the whole
// per-b working set (~74 MB) fits easily in workspace / L2.
// ---------------------------------------------------------------------------
extern "C" void kernel_launch(void* const* d_in, const int* in_sizes, int n_in,
                              void* d_out, int out_size, void* d_ws, size_t ws_size,
                              hipStream_t stream) {
    const float* x      = (const float*)d_in[0];
    const float* ln1_g  = (const float*)d_in[1];
    const float* ln1_b  = (const float*)d_in[2];
    const float* qkv_w  = (const float*)d_in[3];
    const float* qkv_b  = (const float*)d_in[4];
    const float* w_orf  = (const float*)d_in[5];
    const float* proj_w = (const float*)d_in[6];
    const float* proj_b = (const float*)d_in[7];
    const float* ln2_g  = (const float*)d_in[8];
    const float* ln2_b  = (const float*)d_in[9];
    const float* fc1_w  = (const float*)d_in[10];
    const float* fc1_b  = (const float*)d_in[11];
    const float* fc2_w  = (const float*)d_in[12];
    const float* fc2_b  = (const float*)d_in[13];
    float* out = (float*)d_out;

    float* ws = (float*)d_ws;
    size_t o = 0;
    float* ln1buf = ws + o; o += (size_t)SEQ * DIM;           // 1024x768
    float* qkvbuf = ws + o; o += (size_t)SEQ * 3 * DIM;       // 1024x2304
    float* rq     = ws + o; o += (size_t)HEADS * SEQ;
    float* rk     = ws + o; o += (size_t)HEADS * SEQ;
    float* qf     = ws + o; o += (size_t)HEADS * SEQ * MFEAT; // 12x1024x384
    float* kf     = ws + o; o += (size_t)HEADS * SEQ * MFEAT;
    float* ksumb  = ws + o; o += (size_t)HEADS * MFEAT;
    float* dinvb  = ws + o; o += (size_t)HEADS * SEQ;
    float* ktvb   = ws + o; o += (size_t)HEADS * MFEAT * HEAD_DIM;
    float* attnb  = ws + o; o += (size_t)SEQ * DIM;
    float* hb     = ws + o; o += (size_t)SEQ * DIM;
    float* ln2buf = ws + o; o += (size_t)SEQ * DIM;
    float* fc1buf = ws + o; o += (size_t)SEQ * HIDDEN;        // 1024x3072

    const dim3 blk(128);
    const long HZ = (long)SEQ * MFEAT;          // per-head qf/kf stride

    for (int b = 0; b < BATCH; ++b) {
        const float* xb = x + (size_t)b * SEQ * DIM;
        float* outb = out + (size_t)b * SEQ * DIM;

        // LN1
        ln_kernel<<<SEQ, 256, 0, stream>>>(xb, ln1_g, ln1_b, ln1buf);

        // QKV: (1024x768) @ (768x2304) + bias
        wmma_gemm<0><<<dim3(3 * DIM / BN, SEQ / BM, 1), blk, 0, stream>>>(
            ln1buf, qkv_w, qkv_b, qkvbuf, nullptr,
            DIM, DIM, 1, 0, 3 * DIM, 0, 3 * DIM, 0, 0, 0, 0.f, 0.f);

        // 0.5*||x*scale||^2 per (head, token) for q and k
        rowsq_kernel<<<HEADS * SEQ / 256, 256, 0, stream>>>(qkvbuf, rq, rk);

        // Feature maps: per head z, exp((x@w)*scale - sq)/sqrt(M)
        wmma_gemm<1><<<dim3(MFEAT / BN, SEQ / BM, HEADS), blk, 0, stream>>>(
            qkvbuf, w_orf, nullptr, qf, rq,
            HEAD_DIM, 3 * DIM, 1, HEAD_DIM, MFEAT, (long)HEAD_DIM * MFEAT,
            MFEAT, HZ, SEQ, 0, SCALE_F, INV_SQRT_M);
        wmma_gemm<1><<<dim3(MFEAT / BN, SEQ / BM, HEADS), blk, 0, stream>>>(
            qkvbuf + DIM, w_orf, nullptr, kf, rk,
            HEAD_DIM, 3 * DIM, 1, HEAD_DIM, MFEAT, (long)HEAD_DIM * MFEAT,
            MFEAT, HZ, SEQ, 0, SCALE_F, INV_SQRT_M);

        // k_sum and 1/(q.k_sum + eps)
        ksum_kernel<<<HEADS * MFEAT / 256, 256, 0, stream>>>(kf, ksumb);
        dinv_kernel<<<HEADS * SEQ / 256, 256, 0, stream>>>(qf, ksumb, dinvb);

        // ktv[z] = kf[z]^T (384x1024) @ v[z] (1024x64)
        wmma_gemm<0><<<dim3(HEAD_DIM / BN, MFEAT / BM, HEADS), blk, 0, stream>>>(
            kf, qkvbuf + 2 * DIM, nullptr, ktvb, nullptr,
            SEQ, 1, MFEAT, HZ, 3 * DIM, HEAD_DIM,
            HEAD_DIM, (long)MFEAT * HEAD_DIM, 0, 0, 0.f, 0.f);

        // out[z] = (qf[z] (1024x384) @ ktv[z] (384x64)) * dinv, heads -> 768
        wmma_gemm<2><<<dim3(HEAD_DIM / BN, SEQ / BM, HEADS), blk, 0, stream>>>(
            qf, ktvb, nullptr, attnb, dinvb,
            MFEAT, MFEAT, 1, HZ, HEAD_DIM, (long)MFEAT * HEAD_DIM,
            DIM, HEAD_DIM, SEQ, 0, 0.f, 0.f);

        // proj + residual(x): h = x + attn @ proj_w + proj_b
        wmma_gemm<4><<<dim3(DIM / BN, SEQ / BM, 1), blk, 0, stream>>>(
            attnb, proj_w, proj_b, hb, xb,
            DIM, DIM, 1, 0, DIM, 0, DIM, 0, 0, DIM, 0.f, 0.f);

        // LN2
        ln_kernel<<<SEQ, 256, 0, stream>>>(hb, ln2_g, ln2_b, ln2buf);

        // FC1 + GELU
        wmma_gemm<3><<<dim3(HIDDEN / BN, SEQ / BM, 1), blk, 0, stream>>>(
            ln2buf, fc1_w, fc1_b, fc1buf, nullptr,
            DIM, DIM, 1, 0, HIDDEN, 0, HIDDEN, 0, 0, 0, 0.f, 0.f);

        // FC2 + residual(h) -> final output
        wmma_gemm<4><<<dim3(DIM / BN, SEQ / BM, 1), blk, 0, stream>>>(
            fc1buf, fc2_w, fc2_b, outb, hb,
            HIDDEN, HIDDEN, 1, 0, DIM, 0, DIM, 0, 0, DIM, 0.f, 0.f);
    }
    (void)in_sizes; (void)n_in; (void)out_size; (void)ws_size;
}
